// Attention_89515708384005
// MI455X (gfx1250) — matrix operand order repaired and checked by reading, not compiled, not run on hardware
//
#include <hip/hip_runtime.h>

#define DIM     256
#define HEADS   8
#define NTOK    49
#define NPAD    64
#define QKV_SCALE 0.17677669529663687f   // 32^-0.5

// LDS layout (bytes)
#define OFF_Q     0        // [64][256] bf16
#define OFF_K     32768    // [64][256] bf16
#define OFF_VT    65536    // [256][64] bf16
#define OFF_XP    98304    // overlay: x [64][256] bf16 (phase A) | p [8][64][64] bf16 (phase B)
#define OFF_STAGE 163840   // [49*256] f32 async staging
#define LDS_TOTAL (163840 + 50176)

typedef __attribute__((ext_vector_type(16))) __bf16        v16bf;
typedef __attribute__((ext_vector_type(8)))  float         v8f;
typedef __attribute__((ext_vector_type(4)))  unsigned int  u32x4;

// GCC-style vector to match the builtin's V4i prototype exactly
typedef int vi4 __attribute__((__vector_size__(16)));
typedef __attribute__((address_space(1))) vi4 as1_vi4;
typedef __attribute__((address_space(3))) vi4 as3_vi4;

#if defined(__AMDGCN__) && \
    __has_builtin(__builtin_amdgcn_global_load_async_to_lds_b128) && \
    __has_builtin(__builtin_amdgcn_s_wait_asynccnt)
#define USE_ASYNC_X 1
#else
#define USE_ASYNC_X 0
#endif

union Frag {
    v16bf v;
    u32x4 q[2];
};

__device__ __forceinline__ unsigned short f2bf(float f) {
    unsigned int u = __float_as_uint(f);
    u += 0x7FFFu + ((u >> 16) & 1u);          // round-to-nearest-even
    return (unsigned short)(u >> 16);
}

#if USE_ASYNC_X
__device__ __forceinline__ as1_vi4* to_as1(const void* p) {
    return (as1_vi4*)(unsigned long long)(uintptr_t)p;
}
__device__ __forceinline__ as3_vi4* to_as3(void* p) {
    return (as3_vi4*)(unsigned)(uintptr_t)p;   // LDS addr = flat[31:0]
}
#endif

// A-fragment (16x32 bf16, M x K), source row-major [row][k], ld in elements.
// ISA layout: lanes 0-15 row M=lane, K = {0..7, 16..23}; lanes 16-31 K = {8..15, 24..31}.
__device__ __forceinline__ v16bf load_a_frag(const unsigned short* base, int ld,
                                             int row0, int k0, int lane) {
    const int hl = lane >> 4;
    const int m  = row0 + (lane & 15);
    const unsigned short* p0 = base + m * ld + k0 + 8 * hl;
    Frag f;
    f.q[0] = *(const u32x4*)p0;          // K = k0+8*hl .. +7
    f.q[1] = *(const u32x4*)(p0 + 16);   // K = k0+16+8*hl .. +7
    return f.v;
}

// B-fragment (32x16 bf16, K x N), source "transposed" row-major [n][k], ld in elements.
// Lanes 0-15 column N=lane hold K = {0..7, 16..23}; lanes 16-31 K = {8..15, 24..31} (matches A).
__device__ __forceinline__ v16bf load_b_frag(const unsigned short* base, int ld,
                                             int col0, int k0, int lane) {
    const int hl = lane >> 4;
    const int n  = col0 + (lane & 15);
    const unsigned short* p0 = base + n * ld + k0 + 8 * hl;
    Frag f;
    f.q[0] = *(const u32x4*)p0;
    f.q[1] = *(const u32x4*)(p0 + 16);
    return f.v;
}

__device__ __forceinline__ v8f wmma_bf16(v16bf a, v16bf b, v8f c) {
    return __builtin_amdgcn_wmma_f32_16x16x32_bf16(false, a, false, b,
                                                   (short)0, c, false, false);
}

// ---------------------------------------------------------------------------
// Kernel 0: dynamic-position-bias MLP (169x2 -> 16 -> 16 -> 16 -> 8) and
// expansion through rel_idx into a padded [8][64][64] f32 bias table.
// Padded columns get -1e30 so they vanish in softmax.
// ---------------------------------------------------------------------------
__global__ void pos_bias_kernel(
    const float* __restrict__ biases, const float* __restrict__ pw,
    const float* __restrict__ pb,
    const float* __restrict__ g1, const float* __restrict__ be1,
    const float* __restrict__ w1, const float* __restrict__ b1,
    const float* __restrict__ g2, const float* __restrict__ be2,
    const float* __restrict__ w2, const float* __restrict__ b2,
    const float* __restrict__ g3, const float* __restrict__ be3,
    const float* __restrict__ w3, const float* __restrict__ b3,
    const int* __restrict__ rel_idx, float* __restrict__ bias_tab)
{
    __shared__ float p[169][8];
    const int i = threadIdx.x;
    if (i < 169) {
        float h0[16], h1[16];
        const float bx = biases[i * 2 + 0], by = biases[i * 2 + 1];
        for (int j = 0; j < 16; ++j) h0[j] = bx * pw[j] + by * pw[16 + j] + pb[j];

        auto lnrelu = [&](const float* src, const float* g, const float* be, float* dst) {
            float m = 0.f;
            for (int j = 0; j < 16; ++j) m += src[j];
            m *= (1.f / 16.f);
            float v = 0.f;
            for (int j = 0; j < 16; ++j) { float d = src[j] - m; v += d * d; }
            v *= (1.f / 16.f);
            const float inv = rsqrtf(v + 1e-5f);
            for (int j = 0; j < 16; ++j) {
                float t = (src[j] - m) * inv * g[j] + be[j];
                dst[j] = t > 0.f ? t : 0.f;
            }
        };

        lnrelu(h0, g1, be1, h1);
        for (int j = 0; j < 16; ++j) {
            float s = b1[j];
            for (int k = 0; k < 16; ++k) s += h1[k] * w1[k * 16 + j];
            h0[j] = s;
        }
        lnrelu(h0, g2, be2, h1);
        for (int j = 0; j < 16; ++j) {
            float s = b2[j];
            for (int k = 0; k < 16; ++k) s += h1[k] * w2[k * 16 + j];
            h0[j] = s;
        }
        lnrelu(h0, g3, be3, h1);
        for (int j = 0; j < 8; ++j) {
            float s = b3[j];
            for (int k = 0; k < 16; ++k) s += h1[k] * w3[k * 8 + j];
            p[i][j] = s;
        }
    }
    __syncthreads();
    for (int idx = threadIdx.x; idx < HEADS * NPAD * NPAD; idx += blockDim.x) {
        const int hh = idx >> 12;
        const int n  = (idx >> 6) & 63;
        const int m  = idx & 63;
        bias_tab[idx] = (n < NTOK && m < NTOK) ? p[rel_idx[n * NTOK + m]][hh] : -1e30f;
    }
}

// ---------------------------------------------------------------------------
// Kernel 1: transpose + bf16-convert weights: qkv_wT[768][256], proj_wT[256][256]
// (K contiguous per output column so B fragments are two 16B loads).
// ---------------------------------------------------------------------------
__global__ void convert_w_kernel(const float* __restrict__ qkv_w,
                                 const float* __restrict__ proj_w,
                                 unsigned short* __restrict__ qkv_wT,
                                 unsigned short* __restrict__ proj_wT)
{
    const int i = blockIdx.x * blockDim.x + threadIdx.x;
    if (i < 768 * 256) {
        const int n = i >> 8, k = i & 255;
        qkv_wT[i] = f2bf(qkv_w[k * 768 + n]);
    }
    if (i < 256 * 256) {
        const int n = i >> 8, k = i & 255;
        proj_wT[i] = f2bf(proj_w[k * 256 + n]);
    }
}

// ---------------------------------------------------------------------------
// Kernel 2: fully fused per-window attention. One block = one window (batch b).
// 256 threads = 8 wave32; phase B assigns one wave per head.
// ---------------------------------------------------------------------------
__global__ __launch_bounds__(256) void fused_attn_kernel(
    const float* __restrict__ x, const float* __restrict__ qkv_b,
    const float* __restrict__ proj_b,
    const unsigned short* __restrict__ qkv_wT,
    const unsigned short* __restrict__ proj_wT,
    const float* __restrict__ bias_tab, float* __restrict__ out)
{
    extern __shared__ char smem[];
    unsigned short* lds_q  = (unsigned short*)(smem + OFF_Q);   // [64][256] bf16
    unsigned short* lds_k  = (unsigned short*)(smem + OFF_K);   // [64][256] bf16
    unsigned short* lds_vT = (unsigned short*)(smem + OFF_VT);  // [256][64] bf16
    unsigned short* lds_x  = (unsigned short*)(smem + OFF_XP);  // [64][256] bf16 (phase A)
    unsigned short* lds_p  = (unsigned short*)(smem + OFF_XP);  // [8][64][64] bf16 (phase B)

    const int b    = blockIdx.x;
    const int tid  = threadIdx.x;
    const int wave = tid >> 5;
    const int lane = tid & 31;
    const int hl   = lane >> 4;
    const int l15  = lane & 15;

    // ---- stage x tile into LDS as bf16, zero-pad rows 49..63 ----
    const float* xb = x + (size_t)b * NTOK * DIM;
#if USE_ASYNC_X
    {
        // async-copy raw f32 tile (49*256*4 = 50176 B) straight into LDS
        const char* gsrc = (const char*)xb;
        char*       lstg = smem + OFF_STAGE;
        for (int off = tid * 16; off < NTOK * DIM * 4; off += 256 * 16) {
            __builtin_amdgcn_global_load_async_to_lds_b128(
                to_as1(gsrc + off), to_as3(lstg + off), 0, 0);
        }
        __builtin_amdgcn_s_wait_asynccnt(0);
        __syncthreads();
        const float* stage = (const float*)(smem + OFF_STAGE);
        for (int i = tid; i < NPAD * DIM; i += 256) {
            const int row = i >> 8;
            const float v = (row < NTOK) ? stage[i] : 0.0f;
            lds_x[i] = f2bf(v);
        }
    }
#else
    for (int i = tid; i < NPAD * DIM; i += 256) {
        const int row = i >> 8;
        const float v = (row < NTOK) ? xb[row * DIM + (i & 255)] : 0.0f;
        lds_x[i] = f2bf(v);
    }
#endif
    __syncthreads();

    // ---- phase A: qkv = x @ qkv_w + qkv_b ; scatter into q / k / vT ----
    // wave w owns output columns [w*96, w*96+96) of the 768-wide result.
    for (int mt = 0; mt < 4; ++mt) {
        v8f acc[6] = {};
        for (int ks = 0; ks < 8; ++ks) {
            const v16bf a = load_a_frag(lds_x, DIM, mt * 16, ks * 32, lane);
#pragma unroll
            for (int t = 0; t < 6; ++t) {
                const v16bf bb = load_b_frag(qkv_wT, DIM, wave * 96 + t * 16, ks * 32, lane);
                acc[t] = wmma_bf16(a, bb, acc[t]);
            }
        }
#pragma unroll
        for (int t = 0; t < 6; ++t) {
            const int col = wave * 96 + t * 16 + l15;
            const float cb = qkv_b[col];
#pragma unroll
            for (int r = 0; r < 8; ++r) {
                const int row = mt * 16 + r + 8 * hl;
                const float v = acc[t][r] + cb;
                if (col < 256)      lds_q[row * DIM + col]            = f2bf(v * QKV_SCALE);
                else if (col < 512) lds_k[row * DIM + (col - 256)]    = f2bf(v);
                else                lds_vT[(col - 512) * NPAD + row]  = f2bf(v);
            }
        }
    }
    __syncthreads();

    // ---- phase B: per-head attention (wave == head) ----
    const int h = wave;
    v16bf bk[4];
#pragma unroll
    for (int nt = 0; nt < 4; ++nt)
        bk[nt] = load_b_frag(lds_k, DIM, nt * 16, h * 32, lane);

    for (int mt = 0; mt < 4; ++mt) {
        v8f acc[4] = {};
        const v16bf a = load_a_frag(lds_q, DIM, mt * 16, h * 32, lane);
#pragma unroll
        for (int nt = 0; nt < 4; ++nt) acc[nt] = wmma_bf16(a, bk[nt], acc[nt]);

        // add relative position bias (padded cols hold -1e30)
#pragma unroll
        for (int nt = 0; nt < 4; ++nt) {
            const int col = nt * 16 + l15;
#pragma unroll
            for (int r = 0; r < 8; ++r) {
                const int row = mt * 16 + r + 8 * hl;
                acc[nt][r] = acc[nt][r] + bias_tab[h * 4096 + row * 64 + col];
            }
        }

        // softmax over 64 columns; each half-wave (16 lanes) holds one row per VGPR r
#pragma unroll
        for (int r = 0; r < 8; ++r) {
            float v0 = acc[0][r], v1 = acc[1][r], v2 = acc[2][r], v3 = acc[3][r];
            float mx = fmaxf(fmaxf(v0, v1), fmaxf(v2, v3));
            for (int off = 1; off < 16; off <<= 1)
                mx = fmaxf(mx, __shfl_xor(mx, off, 32));
            v0 = __expf(v0 - mx); v1 = __expf(v1 - mx);
            v2 = __expf(v2 - mx); v3 = __expf(v3 - mx);
            float s = v0 + v1 + v2 + v3;
            for (int off = 1; off < 16; off <<= 1)
                s += __shfl_xor(s, off, 32);
            const float inv = 1.0f / s;
            const int row = mt * 16 + r + 8 * hl;
            unsigned short* pr = lds_p + h * 4096 + row * 64 + l15;
            pr[0]  = f2bf(v0 * inv);
            pr[16] = f2bf(v1 * inv);
            pr[32] = f2bf(v2 * inv);
            pr[48] = f2bf(v3 * inv);
        }
    }

    // P @ V : own-head LDS only (no barrier needed), write attn-out back into
    // this head's disjoint column slice of lds_q.
    for (int mt = 0; mt < 4; ++mt) {
        v8f acc[2] = {};
        for (int ks = 0; ks < 2; ++ks) {
            const v16bf a = load_a_frag(lds_p + h * 4096, NPAD, mt * 16, ks * 32, lane);
#pragma unroll
            for (int nt = 0; nt < 2; ++nt) {
                const v16bf bb = load_b_frag(lds_vT, NPAD, h * 32 + nt * 16, ks * 32, lane);
                acc[nt] = wmma_bf16(a, bb, acc[nt]);
            }
        }
#pragma unroll
        for (int nt = 0; nt < 2; ++nt) {
            const int col = h * 32 + nt * 16 + l15;
#pragma unroll
            for (int r = 0; r < 8; ++r) {
                const int row = mt * 16 + r + 8 * hl;
                lds_q[row * DIM + col] = f2bf(acc[nt][r]);
            }
        }
    }
    __syncthreads();

    // ---- phase C: out = attn_out @ proj_w + proj_b ----
    // wave w owns output columns [w*32, w*32+32).
    for (int mt = 0; mt < 4; ++mt) {
        v8f acc[2] = {};
        for (int ks = 0; ks < 8; ++ks) {
            const v16bf a = load_a_frag(lds_q, DIM, mt * 16, ks * 32, lane);
#pragma unroll
            for (int nt = 0; nt < 2; ++nt) {
                const v16bf bb = load_b_frag(proj_wT, DIM, wave * 32 + nt * 16, ks * 32, lane);
                acc[nt] = wmma_bf16(a, bb, acc[nt]);
            }
        }
#pragma unroll
        for (int nt = 0; nt < 2; ++nt) {
            const int col = wave * 32 + nt * 16 + l15;
            const float pb = proj_b[col];
#pragma unroll
            for (int r = 0; r < 8; ++r) {
                const int row = mt * 16 + r + 8 * hl;
                if (row < NTOK)
                    out[(size_t)b * NTOK * DIM + row * DIM + col] = acc[nt][r] + pb;
            }
        }
    }
}

// ---------------------------------------------------------------------------
extern "C" void kernel_launch(void* const* d_in, const int* in_sizes, int n_in,
                              void* d_out, int out_size, void* d_ws, size_t ws_size,
                              hipStream_t stream)
{
    const float* x       = (const float*)d_in[0];
    const float* qkv_w   = (const float*)d_in[1];
    const float* qkv_b   = (const float*)d_in[2];
    const float* proj_w  = (const float*)d_in[3];
    const float* proj_b  = (const float*)d_in[4];
    const float* ppw     = (const float*)d_in[5];
    const float* ppb     = (const float*)d_in[6];
    const float* ln1_g   = (const float*)d_in[7];
    const float* ln1_b   = (const float*)d_in[8];
    const float* w1      = (const float*)d_in[9];
    const float* b1      = (const float*)d_in[10];
    const float* ln2_g   = (const float*)d_in[11];
    const float* ln2_b   = (const float*)d_in[12];
    const float* w2      = (const float*)d_in[13];
    const float* b2      = (const float*)d_in[14];
    const float* ln3_g   = (const float*)d_in[15];
    const float* ln3_b   = (const float*)d_in[16];
    const float* w3      = (const float*)d_in[17];
    const float* b3      = (const float*)d_in[18];
    const float* biases  = (const float*)d_in[19];
    const int*   rel_idx = (const int*)d_in[20];
    // d_in[21], d_in[22] = H, W (hardcoded 7x7)

    float*          bias_tab = (float*)d_ws;                             // 131072 B
    unsigned short* qkv_wT   = (unsigned short*)((char*)d_ws + 131072);  // 393216 B
    unsigned short* proj_wT  = (unsigned short*)((char*)d_ws + 524288);  // 131072 B

    pos_bias_kernel<<<1, 256, 0, stream>>>(biases, ppw, ppb,
                                           ln1_g, ln1_b, w1, b1,
                                           ln2_g, ln2_b, w2, b2,
                                           ln3_g, ln3_b, w3, b3,
                                           rel_idx, bias_tab);
    convert_w_kernel<<<768, 256, 0, stream>>>(qkv_w, proj_w, qkv_wT, proj_wT);

    fused_attn_kernel<<<4096, 256, LDS_TOTAL, stream>>>(
        x, qkv_b, proj_b, qkv_wT, proj_wT, bias_tab, (float*)d_out);
}
